// VectorQuantizer_56831007260897
// MI455X (gfx1250) — matrix operand order repaired and checked by reading, not compiled, not run on hardware
//
#include <hip/hip_runtime.h>
#include <cfloat>

// Problem constants (from the reference)
#define NN     65536
#define DD     256
#define KK     1024
#define BETA_F 0.25f

// gfx1250 async global->LDS copy path (ASYNCcnt), if the toolchain exposes it.
#if defined(__has_builtin)
#if __has_builtin(__builtin_amdgcn_global_load_async_to_lds_b128) && \
    __has_builtin(__builtin_amdgcn_s_wait_asynccnt)
#define USE_ASYNC_LDS 1
#endif
#endif

typedef __attribute__((ext_vector_type(16))) __bf16 v16bf;
typedef __attribute__((ext_vector_type(8)))  float  v8f;

#if defined(USE_ASYNC_LDS)
// Builtin signature (from hipcc diagnostic): first param is
//   '__attribute__((__vector_size__(4*sizeof(int)))) int __device__ *'
// i.e. non-const v4i in the global address space; LDS side is v4i in AS(3).
typedef int v4i_async __attribute__((vector_size(16)));
typedef __attribute__((address_space(1))) v4i_async* g_v4i_ptr;
typedef __attribute__((address_space(3))) v4i_async* l_v4i_ptr;
#endif

union Frag16 {
    uint4 q[2];
    v16bf v;
};

__device__ __forceinline__ unsigned short f2bf(float f) {
    union { float f; unsigned u; } c;
    c.f = f;
    unsigned u = c.u;
    u += 0x7FFFu + ((u >> 16) & 1u);   // round-to-nearest-even
    return (unsigned short)(u >> 16);
}

// ---------------------------------------------------------------------------
// Prep: codebook f32 -> bf16 (workspace), per-code squared norms, zero loss.
// One block per code row (256 threads = 256 dims).
// ---------------------------------------------------------------------------
__global__ __launch_bounds__(256) void vq_prep(const float* __restrict__ E,
                                               unsigned short* __restrict__ ebf,
                                               float* __restrict__ esq,
                                               float* __restrict__ out_loss) {
    const int b = blockIdx.x;   // code index 0..1023
    const int t = threadIdx.x;  // dim 0..255
    float v = E[b * DD + t];
    ebf[b * DD + t] = f2bf(v);

    __shared__ float red[256];
    red[t] = v * v;
    __syncthreads();
    for (int s = 128; s > 0; s >>= 1) {
        if (t < s) red[t] += red[t + s];
        __syncthreads();
    }
    if (t == 0) {
        esq[b] = red[0];
        if (b == 0) *out_loss = 0.0f;   // d_out is poisoned; zero accumulator
    }
}

// ---------------------------------------------------------------------------
// Main: per wave, a 16-row z tile vs all 1024 codes via bf16 WMMA.
// The block double-buffers 16-code B chunks in LDS; staging uses the gfx1250
// async global->LDS engine (ASYNCcnt) when available, overlapped with the
// WMMA chain. Block = 128 threads = 4 waves -> 64 rows per block, 1024 blocks.
// ---------------------------------------------------------------------------
__global__ __launch_bounds__(128) void vq_main(const float* __restrict__ z,
                                               const float* __restrict__ E,
                                               const unsigned short* __restrict__ ebf,
                                               const float* __restrict__ esq,
                                               float* __restrict__ out_zq,
                                               float* __restrict__ out_idx,
                                               float* __restrict__ out_loss) {
    __shared__ unsigned short sA[4][16 * DD];  // 4 waves x (16 rows x 256 dims) bf16 = 32 KB
    __shared__ unsigned short sB[2][16 * DD];  // double-buffered 16-code chunk    = 16 KB
    __shared__ int   sIdx[4][16];
    __shared__ float sRed[128];

    const int tid  = threadIdx.x;
    const int wave = tid >> 5;
    const int lane = tid & 31;
    const int half = lane >> 4;    // 0: lanes 0-15, 1: lanes 16-31
    const int lcol = lane & 15;
    const int rowbase = blockIdx.x * 64 + wave * 16;

    // ---- Stage 16 rows of z as bf16 in this wave's LDS region --------------
    {
        const float4* zsrc = (const float4*)(z + (size_t)rowbase * DD); // 1024 float4
        unsigned short* dst = sA[wave];
        #pragma unroll 4
        for (int j = 0; j < 32; ++j) {
            int e4 = lane + j * 32;          // 0..1023
            float4 v = zsrc[e4];
            union { unsigned short s[4]; uint2 u; } p;
            p.s[0] = f2bf(v.x); p.s[1] = f2bf(v.y);
            p.s[2] = f2bf(v.z); p.s[3] = f2bf(v.w);
            *(uint2*)(dst + e4 * 4) = p.u;   // row-major preserved
        }
    }

    // ---- Stage B chunk 0 (8 KB contiguous = 512 uint4 per chunk) -----------
    const uint4* bsrc = (const uint4*)ebf;   // chunk c at uint4 index c*128
#if defined(USE_ASYNC_LDS)
    {
        g_v4i_ptr g = (g_v4i_ptr)(bsrc + tid);
        l_v4i_ptr l = (l_v4i_ptr)((uint4*)sB[0] + tid);
        __builtin_amdgcn_global_load_async_to_lds_b128(g,       l,       0, 0);
        __builtin_amdgcn_global_load_async_to_lds_b128(g + 128, l + 128, 0, 0);
        __builtin_amdgcn_global_load_async_to_lds_b128(g + 256, l + 256, 0, 0);
        __builtin_amdgcn_global_load_async_to_lds_b128(g + 384, l + 384, 0, 0);
        __builtin_amdgcn_s_wait_asynccnt(0);
    }
#else
    {
        uint4* dst = (uint4*)sB[0];
        #pragma unroll
        for (int j = 0; j < 4; ++j)
            dst[tid + j * 128] = bsrc[tid + j * 128];
    }
#endif
    __syncthreads();

    float bval[8];
    int   bidx[8];
    #pragma unroll
    for (int r = 0; r < 8; ++r) { bval[r] = FLT_MAX; bidx[r] = 0; }

    // A fragment base: row = lane%16, first k segment at k = (lane/16)*8
    const unsigned short* aBase = sA[wave] + lcol * DD + half * 8;

    // ---- Sweep all 1024 codes in chunks of 16 columns ----------------------
    for (int chunk = 0; chunk < KK / 16; ++chunk) {
        const int nextc = (chunk + 1) & (KK / 16 - 1);
        const uint4* psrc = bsrc + (size_t)nextc * 128;

#if defined(USE_ASYNC_LDS)
        // Kick off async staging of the next chunk straight into LDS; it runs
        // behind the WMMA chain (tracked by ASYNCcnt, no VGPRs involved).
        {
            g_v4i_ptr g = (g_v4i_ptr)(psrc + tid);
            l_v4i_ptr l = (l_v4i_ptr)((uint4*)sB[(chunk + 1) & 1] + tid);
            __builtin_amdgcn_global_load_async_to_lds_b128(g,       l,       0, 0);
            __builtin_amdgcn_global_load_async_to_lds_b128(g + 128, l + 128, 0, 0);
            __builtin_amdgcn_global_load_async_to_lds_b128(g + 256, l + 256, 0, 0);
            __builtin_amdgcn_global_load_async_to_lds_b128(g + 384, l + 384, 0, 0);
        }
#else
        // Fallback: prefetch next chunk into registers, commit after compute.
        uint4 pf0 = psrc[tid];
        uint4 pf1 = psrc[tid + 128];
        uint4 pf2 = psrc[tid + 256];
        uint4 pf3 = psrc[tid + 384];
#endif

        const int code = chunk * 16 + lcol;                // this lane's column
        const float en = esq[code];
        const unsigned short* bBase = sB[chunk & 1] + lcol * DD + half * 8;

        // Two accumulators (even/odd kt) break the serial WMMA RAW chain.
        v8f c0 = {0.f, 0.f, 0.f, 0.f, 0.f, 0.f, 0.f, 0.f};
        v8f c1 = {0.f, 0.f, 0.f, 0.f, 0.f, 0.f, 0.f, 0.f};
        #pragma unroll
        for (int kt = 0; kt < 8; kt += 2) {                // depth 256 = 8 x 32
            Frag16 a0, b0, a1, b1;
            const unsigned short* ap0 = aBase + kt * 32;
            const unsigned short* bp0 = bBase + kt * 32;
            const unsigned short* ap1 = aBase + (kt + 1) * 32;
            const unsigned short* bp1 = bBase + (kt + 1) * 32;
            a0.q[0] = *(const uint4*)(ap0);
            a0.q[1] = *(const uint4*)(ap0 + 16);
            b0.q[0] = *(const uint4*)(bp0);
            b0.q[1] = *(const uint4*)(bp0 + 16);
            a1.q[0] = *(const uint4*)(ap1);
            a1.q[1] = *(const uint4*)(ap1 + 16);
            b1.q[0] = *(const uint4*)(bp1);
            b1.q[1] = *(const uint4*)(bp1 + 16);
            c0 = __builtin_amdgcn_wmma_f32_16x16x32_bf16(
                     false, a0.v, false, b0.v, (short)0, c0, false, false);
            c1 = __builtin_amdgcn_wmma_f32_16x16x32_bf16(
                     false, a1.v, false, b1.v, (short)0, c1, false, false);
        }

        #pragma unroll
        for (int r = 0; r < 8; ++r) {
            float sc = en - 2.0f * (c0[r] + c1[r]);        // ||e||^2 - 2 z.e
            if (sc < bval[r] || (sc == bval[r] && code < bidx[r])) {
                bval[r] = sc; bidx[r] = code;
            }
        }

#if defined(USE_ASYNC_LDS)
        __builtin_amdgcn_s_wait_asynccnt(0);   // my async writes are in LDS
#else
        uint4* dst = (uint4*)sB[(chunk + 1) & 1];
        dst[tid]       = pf0;
        dst[tid + 128] = pf1;
        dst[tid + 256] = pf2;
        dst[tid + 384] = pf3;
#endif
        __syncthreads();                       // everyone's writes visible
    }

    // ---- Argmin across the 16 columns held in each half-wave ---------------
    #pragma unroll
    for (int off = 1; off < 16; off <<= 1) {
        #pragma unroll
        for (int r = 0; r < 8; ++r) {
            float ov = __shfl_xor(bval[r], off, 32);
            int   oi = __shfl_xor(bidx[r], off, 32);
            if (ov < bval[r] || (ov == bval[r] && oi < bidx[r])) {
                bval[r] = ov; bidx[r] = oi;
            }
        }
    }
    // C layout: VGPR r <-> row r + 8*half, so lane lcol==0 of each half owns
    // rows 0..7 (half 0) and 8..15 (half 1).
    if (lcol == 0) {
        #pragma unroll
        for (int r = 0; r < 8; ++r) {
            int rloc = r + 8 * half;
            sIdx[wave][rloc] = bidx[r];
            out_idx[rowbase + rloc] = (float)bidx[r];
        }
    }
    __syncthreads();

    // ---- Epilogue: gather z_q in f32, accumulate (z_q - z)^2 ---------------
    float acc = 0.0f;
    for (int r = 0; r < 16; ++r) {
        const int row = rowbase + r;
        const int idx = sIdx[wave][r];
        const float4* ep = (const float4*)(E + (size_t)idx * DD) + lane * 2;
        const float4* zp = (const float4*)(z + (size_t)row * DD) + lane * 2;
        float4*       op = (float4*)(out_zq + (size_t)row * DD) + lane * 2;
        #pragma unroll
        for (int j = 0; j < 2; ++j) {
            float4 e4 = ep[j], z4 = zp[j];
            float dx = e4.x - z4.x, dy = e4.y - z4.y;
            float dz = e4.z - z4.z, dw = e4.w - z4.w;
            acc += dx * dx + dy * dy + dz * dz + dw * dw;
            op[j] = e4;   // z_q_st == z_q numerically
        }
    }
    sRed[tid] = acc;
    __syncthreads();
    for (int s = 64; s > 0; s >>= 1) {
        if (tid < s) sRed[tid] += sRed[tid + s];
        __syncthreads();
    }
    if (tid == 0) {
        // loss = (1+beta) * mean((z_q - z)^2) over N*D elements
        atomicAdd(out_loss, sRed[0] * ((1.0f + BETA_F) / ((float)NN * (float)DD)));
    }
}

// ---------------------------------------------------------------------------
extern "C" void kernel_launch(void* const* d_in, const int* in_sizes, int n_in,
                              void* d_out, int out_size, void* d_ws, size_t ws_size,
                              hipStream_t stream) {
    const float* z = (const float*)d_in[0];             // [N,1,256] f32
    const float* E = (const float*)d_in[1];             // [K,256]   f32

    // workspace: bf16 codebook (512 KB) + per-code squared norms (4 KB)
    unsigned short* ebf = (unsigned short*)d_ws;
    float* esq = (float*)((char*)d_ws + (size_t)KK * DD * sizeof(unsigned short));

    float* out      = (float*)d_out;
    float* out_zq   = out;                               // N*256 floats
    float* out_idx  = out + (size_t)NN * DD;             // N floats (indices as float)
    float* out_loss = out + (size_t)NN * DD + NN;        // 1 float

    vq_prep<<<KK, DD, 0, stream>>>(E, ebf, esq, out_loss);
    vq_main<<<NN / 64, 128, 0, stream>>>(z, E, ebf, esq, out_zq, out_idx, out_loss);
}
